// TrajectoryHGNN_65652870087171
// MI455X (gfx1250) — matrix operand order repaired and checked by reading, not compiled, not run on hardware
//
#include <hip/hip_runtime.h>

// Problem constants (from reference)
#define B_   64
#define N_   512
#define T_   8
#define DIN_ 2
#define H_   256
#define L_   3
#define E_   8192
#define M_   128
#define P2_  24          // P*2
#define R_   (B_*N_)     // 32768 rows
#define EPS_ 1e-5f
#define CH_  32          // channel chunk held in LDS per conv workgroup

typedef __attribute__((ext_vector_type(16))) __bf16 v16bf;
typedef __attribute__((ext_vector_type(8)))  __bf16 v8bf;
typedef __attribute__((ext_vector_type(8)))  float  v8f;

// ---------------------------------------------------------------------------
// 1) thetas -> bf16, transposed: TbT[l][n][k] = theta[l][k][n]
//    (makes WMMA B-operand loads contiguous in K per lane)
// ---------------------------------------------------------------------------
__global__ __launch_bounds__(256) void k_theta_T(const float* __restrict__ th,
                                                 __bf16* __restrict__ tbt) {
    int t = blockIdx.x * 256 + threadIdx.x;
    if (t >= L_ * H_ * H_) return;
    int l   = t / (H_ * H_);
    int rem = t - l * (H_ * H_);
    int k   = rem / H_;
    int n   = rem - k * H_;
    tbt[l * H_ * H_ + n * H_ + k] = (__bf16)th[t];
}

// ---------------------------------------------------------------------------
// 2) temporal encoder: h[row][c] = mean_t relu(x0*W0 + x1*W1 + b)
//    one block per (b,n) row, 256 threads = 256 channels
// ---------------------------------------------------------------------------
__global__ __launch_bounds__(256) void k_encoder(const float* __restrict__ obs,
                                                 const float* __restrict__ Wenc,
                                                 const float* __restrict__ benc,
                                                 float* __restrict__ h,
                                                 __bf16* __restrict__ hb) {
    int row = blockIdx.x;         // b*N + n
    int c   = threadIdx.x;
    const float* x = obs + (size_t)row * T_ * DIN_;
    float w0 = Wenc[c], w1 = Wenc[H_ + c], bb = benc[c];
    float acc = 0.f;
#pragma unroll
    for (int t = 0; t < T_; ++t) {
        float v = x[t * 2] * w0 + x[t * 2 + 1] * w1 + bb;
        acc += fmaxf(v, 0.f);
    }
    acc *= (1.0f / T_);
    h [(size_t)row * H_ + c] = acc;
    hb[(size_t)row * H_ + c] = (__bf16)acc;
}

// ---------------------------------------------------------------------------
// 3) GEMM: Y[32768,256] = X(bf16)[32768,256] @ theta, theta given transposed
//    bf16 (Bt[n][k]).  One wave -> 16x64 tile, 4 f32 accumulators,
//    K-loop 8 steps of v_wmma_f32_16x16x32_bf16.
// ---------------------------------------------------------------------------
#define BTILE(ACC, NOFF)                                                        \
    {                                                                           \
        const __bf16* br = Bt + (size_t)(nb + (NOFF) + mr) * H_ + k0 + 8 * g;   \
        v8bf b0 = *(const v8bf*)br;                                             \
        v8bf b1 = *(const v8bf*)(br + 16);                                      \
        v16bf bm = __builtin_shufflevector(b0, b1, 0, 1, 2, 3, 4, 5, 6, 7,      \
                                           8, 9, 10, 11, 12, 13, 14, 15);       \
        ACC = __builtin_amdgcn_wmma_f32_16x16x32_bf16(false, a, false, bm,      \
                                                      (short)0, ACC, false,     \
                                                      false);                   \
    }

__global__ __launch_bounds__(256) void k_gemm(const __bf16* __restrict__ X,
                                              const __bf16* __restrict__ Bt,
                                              float* __restrict__ Y) {
    int wave = threadIdx.x >> 5;
    int lane = threadIdx.x & 31;
    int w    = blockIdx.x * 8 + wave;
    int rowTile = w >> 2;                 // 2048 row tiles
    int colGrp  = w & 3;                  // 4 groups of 64 cols
    int m0 = rowTile * 16;
    int nb = colGrp * 64;
    int g  = lane >> 4;                   // lane half (K split)
    int mr = lane & 15;                   // M (A) / N (B) index

    const __bf16* arow = X + (size_t)(m0 + mr) * H_;
    v8f acc0 = {}, acc1 = {}, acc2 = {}, acc3 = {};

    for (int k0 = 0; k0 < H_; k0 += 32) {
        // A: lane holds K = k0+8g..+7 (elems 0..7) and k0+16+8g..+7 (elems 8..15)
        v8bf alo = *(const v8bf*)(arow + k0 + 8 * g);
        v8bf ahi = *(const v8bf*)(arow + k0 + 16 + 8 * g);
        v16bf a = __builtin_shufflevector(alo, ahi, 0, 1, 2, 3, 4, 5, 6, 7,
                                          8, 9, 10, 11, 12, 13, 14, 15);
        BTILE(acc0, 0)
        BTILE(acc1, 16)
        BTILE(acc2, 32)
        BTILE(acc3, 48)
    }

    // C/D layout: VGPR r -> row m0 + r + 8*g, col nb + mr
#pragma unroll
    for (int r = 0; r < 8; ++r) {
        size_t row = (size_t)(m0 + r + 8 * g) * H_;
        Y[row + nb + 0  + mr] = acc0[r];
        Y[row + nb + 16 + mr] = acc1[r];
        Y[row + nb + 32 + mr] = acc2[r];
        Y[row + nb + 48 + mr] = acc3[r];
    }
}

// ---------------------------------------------------------------------------
// 4) Fused hypergraph conv: scatter(node->edge), edge-degree normalize,
//    scatter(edge->node), node-degree normalize + bias, BatchNorm over node
//    axis, ReLU.  One block per (batch, 32-channel chunk); all accumulation
//    in LDS via ds_add_f32 atomics.
// ---------------------------------------------------------------------------
__global__ __launch_bounds__(256) void k_conv(const float* __restrict__ xt,
                                              const int* __restrict__ idx,
                                              const float* __restrict__ bias,
                                              const float* __restrict__ gamma,
                                              const float* __restrict__ beta,
                                              float* __restrict__ hsoc,
                                              __bf16* __restrict__ hb) {
    __shared__ float msg [M_ * CH_];   // 16 KB
    __shared__ float outb[N_ * CH_];   // 64 KB
    __shared__ float bdeg[M_];
    __shared__ float ddeg[N_];
    __shared__ float psum[CH_ * 8];
    __shared__ float psq [CH_ * 8];
    __shared__ float scl [CH_];
    __shared__ float shf [CH_];

    int b   = blockIdx.x >> 3;            // batch
    int c0  = (blockIdx.x & 7) * CH_;     // channel chunk base
    int tid = threadIdx.x;
    int hc  = tid & (CH_ - 1);            // channel within chunk
    int er  = tid >> 5;                   // entry row 0..7

    for (int i = tid; i < M_ * CH_; i += 256) msg[i]  = 0.f;
    for (int i = tid; i < N_ * CH_; i += 256) outb[i] = 0.f;
    if (tid < M_) bdeg[tid] = 0.f;
    for (int i = tid; i < N_; i += 256) ddeg[i] = 0.f;
    __syncthreads();

    const int* nid = idx + (size_t)b * 2 * E_;
    const int* eid = nid + E_;
    const float* xb = xt + (size_t)b * N_ * H_ + c0;

    // pass 1: msg[edge] += xt[node]; count degrees
    for (int e0 = 0; e0 < E_; e0 += 8) {
        int e    = e0 + er;
        int node = nid[e] & (N_ - 1);
        int edge = eid[e] & (M_ - 1);
        float v  = xb[(size_t)node * H_ + hc];
        atomicAdd(&msg[edge * CH_ + hc], v);
        if (hc == 0) atomicAdd(&bdeg[edge], 1.f);
        if (hc == 1) atomicAdd(&ddeg[node], 1.f);
    }
    __syncthreads();

    // msg /= max(bdeg,1)
    for (int i = tid; i < M_ * CH_; i += 256)
        msg[i] = msg[i] / fmaxf(bdeg[i >> 5], 1.f);
    __syncthreads();

    // pass 2: out[node] += msg[edge]
    for (int e0 = 0; e0 < E_; e0 += 8) {
        int e    = e0 + er;
        int node = nid[e] & (N_ - 1);
        int edge = eid[e] & (M_ - 1);
        atomicAdd(&outb[node * CH_ + hc], msg[edge * CH_ + hc]);
    }
    __syncthreads();

    // out = out / max(ddeg,1) + bias   (i & 31 == hc since 256 % 32 == 0)
    float bi = bias[c0 + hc];
    for (int i = tid; i < N_ * CH_; i += 256)
        outb[i] = outb[i] / fmaxf(ddeg[i >> 5], 1.f) + bi;
    __syncthreads();

    // BatchNorm stats over the 512 nodes, per channel
    float s = 0.f, sq = 0.f;
    for (int n = er; n < N_; n += 8) {
        float v = outb[n * CH_ + hc];
        s += v; sq += v * v;
    }
    psum[hc * 8 + er] = s;
    psq [hc * 8 + er] = sq;
    __syncthreads();
    if (tid < CH_) {
        float ss = 0.f, qq = 0.f;
#pragma unroll
        for (int gph = 0; gph < 8; ++gph) { ss += psum[tid * 8 + gph]; qq += psq[tid * 8 + gph]; }
        float mu  = ss * (1.f / N_);
        float var = qq * (1.f / N_) - mu * mu;
        float sc  = gamma[c0 + tid] * rsqrtf(var + EPS_);
        scl[tid] = sc;
        shf[tid] = beta[c0 + tid] - mu * sc;
    }
    __syncthreads();

    // apply BN + ReLU, write f32 (residual) and bf16 (next GEMM input)
    float*  hrow  = hsoc + (size_t)b * N_ * H_ + c0;
    __bf16* hbrow = hb   + (size_t)b * N_ * H_ + c0;
    for (int n = er; n < N_; n += 8) {
        float v = outb[n * CH_ + hc] * scl[hc] + shf[hc];
        v = fmaxf(v, 0.f);
        hrow [(size_t)n * H_ + hc] = v;
        hbrow[(size_t)n * H_ + hc] = (__bf16)v;
    }
}

// ---------------------------------------------------------------------------
// 5) decoder: out[row][col] = b_dec[col] + sum_k (h+hsoc)[row][k]*Wd[k][col]
// ---------------------------------------------------------------------------
__global__ __launch_bounds__(256) void k_dec(const float* __restrict__ h,
                                             const float* __restrict__ hs,
                                             const float* __restrict__ Wd,
                                             const float* __restrict__ bd,
                                             float* __restrict__ out) {
    int t = blockIdx.x * 256 + threadIdx.x;
    if (t >= R_ * P2_) return;
    int row = t / P2_;
    int col = t - row * P2_;
    const float* hr = h  + (size_t)row * H_;
    const float* sr = hs + (size_t)row * H_;
    float acc = bd[col];
#pragma unroll 4
    for (int k = 0; k < H_; ++k)
        acc += (hr[k] + sr[k]) * Wd[k * P2_ + col];
    out[t] = acc;
}

// ---------------------------------------------------------------------------
extern "C" void kernel_launch(void* const* d_in, const int* in_sizes, int n_in,
                              void* d_out, int out_size, void* d_ws, size_t ws_size,
                              hipStream_t stream) {
    const float* obs    = (const float*)d_in[0];
    const int*   idx    = (const int*)  d_in[1];
    const float* Wenc   = (const float*)d_in[2];
    const float* benc   = (const float*)d_in[3];
    const float* thetas = (const float*)d_in[4];
    const float* cbias  = (const float*)d_in[5];
    const float* gam    = (const float*)d_in[6];
    const float* bet    = (const float*)d_in[7];
    const float* Wdec   = (const float*)d_in[8];
    const float* bdec   = (const float*)d_in[9];

    char* ws = (char*)d_ws;
    size_t o = 0;
    float*  h_enc = (float*) (ws + o); o += (size_t)R_ * H_ * 4;   // 32 MB
    float*  hsoc  = (float*) (ws + o); o += (size_t)R_ * H_ * 4;   // 32 MB
    float*  xt    = (float*) (ws + o); o += (size_t)R_ * H_ * 4;   // 32 MB
    __bf16* hb    = (__bf16*)(ws + o); o += (size_t)R_ * H_ * 2;   // 16 MB
    __bf16* tbt   = (__bf16*)(ws + o); o += (size_t)L_ * H_ * H_ * 2;

    k_theta_T<<<(L_ * H_ * H_ + 255) / 256, 256, 0, stream>>>(thetas, tbt);
    k_encoder<<<R_, 256, 0, stream>>>(obs, Wenc, benc, h_enc, hb);
    for (int l = 0; l < L_; ++l) {
        // 2048 row tiles * 4 col groups / 8 waves per block = 1024 blocks
        k_gemm<<<1024, 256, 0, stream>>>(hb, tbt + (size_t)l * H_ * H_, xt);
        k_conv<<<B_ * 8, 256, 0, stream>>>(xt, idx, cbias + l * H_,
                                           gam + l * H_, bet + l * H_, hsoc, hb);
    }
    k_dec<<<(R_ * P2_ + 255) / 256, 256, 0, stream>>>(h_enc, hsoc, Wdec, bdec,
                                                      (float*)d_out);
}
